// PPGN_LGVR_plus_34119220199797
// MI455X (gfx1250) — matrix operand order repaired and checked by reading, not compiled, not run on hardware
//
#include <hip/hip_runtime.h>
#include <hip/hip_bf16.h>
#include <math.h>

// ---------------------------------------------------------------------------
// Dims
// ---------------------------------------------------------------------------
#define BATCH 8
#define NN    128
#define NSQ   (NN*NN)            // 16384
#define CIN0  17                 // NL+1
#define CH    256                // C
#define FEAT  512                // 2*C
#define PH    16                 // PH_INIT
#define PHI   50                 // PH_INDS
#define K1V   32
#define KE1V  4
#define NPAIR 8128               // 128*127/2
#define SORTN 8192

typedef __attribute__((ext_vector_type(16))) _Float16 v16h;
typedef __attribute__((ext_vector_type(8)))  float    v8f;

// ---------------------------------------------------------------------------
// gfx1250 async global->LDS copy (probe-guarded; full fallback kept)
// Verified signature: (v4i addrspace(1)*, v4i addrspace(3)*, imm off, imm cpol)
// ---------------------------------------------------------------------------
#if defined(__has_builtin)
# if __has_builtin(__builtin_amdgcn_global_load_async_to_lds_b128)
#  define HAVE_ASYNC_COPY 1
# endif
#endif

#ifdef HAVE_ASYNC_COPY
typedef int vi4 __attribute__((vector_size(16)));
typedef __attribute__((address_space(1))) vi4 g_vi4;
typedef __attribute__((address_space(3))) vi4 l_vi4;
__device__ __forceinline__ void async_copy_b128(const float* g, float* l) {
    __builtin_amdgcn_global_load_async_to_lds_b128(
        (g_vi4*)(void*)(const_cast<float*>(g)),
        (l_vi4*)(void*)(l), 0, 0);
}
__device__ __forceinline__ void wait_async0() {
# if __has_builtin(__builtin_amdgcn_s_wait_asynccnt)
    __builtin_amdgcn_s_wait_asynccnt(0);
# else
    asm volatile("s_wait_asynccnt 0" ::: "memory");
# endif
}
#endif

// ---------------------------------------------------------------------------
// Generic WMMA GEMM:  C[z][m][p] = act( sum_k A(k,m) * B(k,p) + bias[m] )
//   AROW=false : A stored (K,M) row-major  (the reference 'W' layout)
//   AROW=true  : A stored (M,K) row-major  (for the einsum m1 matrix)
//   B is split on K: k < Ksplit -> B0, else B1 (for the skip concat)
// Tile: 64(M) x 64(P), 4 waves, K step 32.  Steady state: raw fp32 tiles come
// in via async global->LDS DMA, get converted into f16 *fragment-major* LDS
// (each lane's 16 operand f16 contiguous -> v16h load = 2x ds_load_b128),
// then v_wmma_f32_16x16x32_f16.  Ragged tiles fall back to guarded loads.
// ---------------------------------------------------------------------------
#define TM 64
#define TP 64
#define TK 32

template<bool AROW>
__global__ __launch_bounds__(128)
void wmma_gemm(const float* __restrict__ A,
               const float* __restrict__ B0,
               const float* __restrict__ B1, int Ksplit,
               const float* __restrict__ bias,
               float* __restrict__ C,
               int M, int K, int P,
               long sA, long sB0, long sB1, long sC,
               int act)   // 0 none, 1 relu, 2 sigmoid
{
    // [strip/pt][lane][elem]
    __shared__ __align__(32) _Float16 Af[4][32][16];
    __shared__ __align__(32) _Float16 Bf[4][32][16];
#ifdef HAVE_ASYNC_COPY
    __shared__ __align__(16) float Braw[TK][TP];
    __shared__ __align__(16) float Araw[TM * TK];   // AROW: [mi][kk], else [kk][mi]
#endif

    const int tid  = threadIdx.x;
    const int wave = tid >> 5;
    const int lane = tid & 31;
    const int z    = blockIdx.z;
    const int tileP = blockIdx.x * TP;
    const int tileM = blockIdx.y * TM;

    const float* Ab  = A  + (long)z * sA;
    const float* B0b = B0 + (long)z * sB0;
    const float* B1b = B1 ? (B1 + (long)z * sB1) : (const float*)0;

    v8f acc[4];
#pragma unroll
    for (int i = 0; i < 4; ++i)
#pragma unroll
        for (int e = 0; e < 8; ++e) acc[i][e] = 0.0f;

    for (int k0 = 0; k0 < K; k0 += TK) {
        bool asyncA = false, asyncB = false;
#ifdef HAVE_ASYNC_COPY
        // ---- async DMA of raw fp32 tiles (fully in-bounds tiles only) ----
        if (tileM + TM <= M && k0 + TK <= K) {
            asyncA = true;
            if (AROW) {
#pragma unroll
                for (int r = 0; r < 4; ++r) {
                    int i  = r * 128 + tid;     // 512 chunks of 4 floats
                    int mi = i >> 3;            // 0..63
                    int kq = (i & 7) * 4;       // 0..28
                    async_copy_b128(Ab + (long)(tileM + mi) * K + k0 + kq,
                                    &Araw[mi * TK + kq]);
                }
            } else {
#pragma unroll
                for (int r = 0; r < 4; ++r) {
                    int i  = r * 128 + tid;
                    int ki = i >> 4;            // 0..31
                    int mq = (i & 15) * 4;      // 0..60
                    async_copy_b128(Ab + (long)(k0 + ki) * M + tileM + mq,
                                    &Araw[ki * TM + mq]);
                }
            }
        }
        {
            const float* Bsrc = (const float*)0;
            if (k0 + TK <= Ksplit)                     Bsrc = B0b + (long)k0 * P;
            else if (k0 >= Ksplit && k0 + TK <= K)     Bsrc = B1b + (long)(k0 - Ksplit) * P;
            if (Bsrc && tileP + TP <= P) {
                asyncB = true;
#pragma unroll
                for (int r = 0; r < 4; ++r) {
                    int i  = r * 128 + tid;
                    int ki = i >> 4;            // 0..31
                    int pq = (i & 15) * 4;      // 0..60
                    async_copy_b128(Bsrc + (long)ki * P + tileP + pq, &Braw[ki][pq]);
                }
            }
        }
        if (asyncA || asyncB) wait_async0();
#endif
        // ---- guarded fallback staging (ragged tiles / no async builtin) ----
        if (!asyncA) {
#pragma unroll
            for (int r = 0; r < 8; ++r) {
                int idx = r * 128 + tid;          // 1024 f16-pairs
                int mi  = idx >> 4;               // 0..63
                int kk  = (idx & 15) * 2;         // even 0..30
                int m = tileM + mi, k = k0 + kk;
                float a0 = 0.0f, a1 = 0.0f;
                if (m < M) {
                    if (AROW) {
                        if (k     < K) a0 = Ab[(long)m * K + k];
                        if (k + 1 < K) a1 = Ab[(long)m * K + k + 1];
                    } else {
                        if (k     < K) a0 = Ab[(long)k * M + m];
                        if (k + 1 < K) a1 = Ab[(long)(k + 1) * M + m];
                    }
                }
                int strip = mi >> 4, lrow = mi & 15;
                int half  = (kk >> 3) & 1;
                int j     = (kk & 7) | ((kk >> 4) << 3);   // even
                union { _Float16 h[2]; unsigned u; } pk;
                pk.h[0] = (_Float16)a0; pk.h[1] = (_Float16)a1;
                *(unsigned*)&Af[strip][lrow + 16 * half][j] = pk.u;
            }
        }
        if (!asyncB) {
#pragma unroll
            for (int r = 0; r < 8; ++r) {
                int idx = r * 128 + tid;      // 1024 f16-pairs
                int pp  = idx & 63;
                int ki2 = (idx >> 6) * 2;     // even 0..30
                int k = k0 + ki2, p = tileP + pp;
                float b0v = 0.0f, b1v = 0.0f;
                if (p < P) {
                    if (k < K)
                        b0v = (k < Ksplit) ? B0b[(long)k * P + p]
                                           : B1b[(long)(k - Ksplit) * P + p];
                    if (k + 1 < K)
                        b1v = (k + 1 < Ksplit) ? B0b[(long)(k + 1) * P + p]
                                               : B1b[(long)(k + 1 - Ksplit) * P + p];
                }
                int pt = pp >> 4, col = pp & 15, half = (ki2 >> 3) & 1, j = (ki2 & 7) | ((ki2 >> 4) << 3);
                union { _Float16 h[2]; unsigned u; } pk;
                pk.h[0] = (_Float16)b0v; pk.h[1] = (_Float16)b1v;
                *(unsigned*)&Bf[pt][col + 16 * half][j] = pk.u;
            }
        }
        // prefetch next B slab (global_prefetch_b8)
        {
            int kn = k0 + TK;
            if (kn < K && kn < Ksplit)
                __builtin_prefetch(B0b + (long)kn * P + tileP + (tid & 63), 0, 1);
        }
        __syncthreads();

#ifdef HAVE_ASYNC_COPY
        if (asyncA || asyncB) {                 // uniform conditions -> legal
            if (asyncA) {
#pragma unroll
                for (int r = 0; r < 8; ++r) {
                    int idx = r * 128 + tid;
                    int mi  = idx >> 4;
                    int kk  = (idx & 15) * 2;
                    float a0, a1;
                    if (AROW) { a0 = Araw[mi * TK + kk];  a1 = Araw[mi * TK + kk + 1]; }
                    else      { a0 = Araw[kk * TM + mi];  a1 = Araw[(kk + 1) * TM + mi]; }
                    int strip = mi >> 4, lrow = mi & 15;
                    int half  = (kk >> 3) & 1;
                    int j     = (kk & 7) | ((kk >> 4) << 3);
                    union { _Float16 h[2]; unsigned u; } pk;
                    pk.h[0] = (_Float16)a0; pk.h[1] = (_Float16)a1;
                    *(unsigned*)&Af[strip][lrow + 16 * half][j] = pk.u;
                }
            }
            if (asyncB) {
#pragma unroll
                for (int r = 0; r < 8; ++r) {
                    int idx = r * 128 + tid;
                    int pp  = idx & 63;
                    int ki2 = (idx >> 6) * 2;
                    float b0v = Braw[ki2][pp];
                    float b1v = Braw[ki2 + 1][pp];
                    int pt = pp >> 4, col = pp & 15, half = (ki2 >> 3) & 1, j = (ki2 & 7) | ((ki2 >> 4) << 3);
                    union { _Float16 h[2]; unsigned u; } pk;
                    pk.h[0] = (_Float16)b0v; pk.h[1] = (_Float16)b1v;
                    *(unsigned*)&Bf[pt][col + 16 * half][j] = pk.u;
                }
            }
            __syncthreads();
        }
#endif

        // ---- fragments: one contiguous 32B LDS load each ----
        v16h af = *(const v16h*)(&Af[wave][lane][0]);
#pragma unroll
        for (int pt = 0; pt < 4; ++pt) {
            v16h bf = *(const v16h*)(&Bf[pt][lane][0]);
            acc[pt] = __builtin_amdgcn_wmma_f32_16x16x32_f16(
                false, af, false, bf, (short)0, acc[pt], false, false);
        }
        __syncthreads();
    }

    // ---- epilogue (C/D 16x16 f32 layout) ----
    const int col0  = lane & 15;
    const int rbase = (lane < 16) ? 0 : 8;
#pragma unroll
    for (int pt = 0; pt < 4; ++pt) {
#pragma unroll
        for (int r = 0; r < 8; ++r) {
            int m = tileM + wave * 16 + rbase + r;
            int p = tileP + pt * 16 + col0;
            if (m < M && p < P) {
                float v = acc[pt][r];
                if (bias) v += bias[m];
                if (act == 1)      v = fmaxf(v, 0.0f);
                else if (act == 2) v = 1.0f / (1.0f + __expf(-v));
                C[(long)z * sC + (long)m * P + p] = v;
            }
        }
    }
}

// ---------------------------------------------------------------------------
// Elementwise kernels
// ---------------------------------------------------------------------------
__global__ void diag_kernel(const float* __restrict__ x, float* __restrict__ v) {
    int i = blockIdx.x * 256 + threadIdx.x;
    if (i < BATCH * CH * NN) {
        int n = i & 127; int bc = i >> 7;
        v[i] = x[(long)bc * NSQ + n * (NN + 1)];
    }
}

__global__ void vsp_kernel(float* __restrict__ v, const float* __restrict__ sp,
                           const float* __restrict__ eps) {
    int i = blockIdx.x * 256 + threadIdx.x;
    if (i < BATCH * CH * NN) {
        int n = i & 127; int b = i / (CH * NN);
        v[i] += (1.0f + eps[0]) * sp[b * NN + n];
    }
}

__global__ void build_ef_kernel(const float* __restrict__ v, float* __restrict__ ef) {
    int i = blockIdx.x * 256 + threadIdx.x;
    if (i < BATCH * FEAT * NSQ) {
        int m  = i & 127;
        int n  = (i >> 7) & 127;
        int ch = (i >> 14) & 511;
        int b  = i >> 23;
        int c  = ch & 255;
        float vn = v[(b * CH + c) * NN + n];
        float vm = v[(b * CH + c) * NN + m];
        ef[i] = (ch < CH) ? (vn + vm) : fabsf(vn - vm);
    }
}

__global__ void mask_adj_kernel(float* __restrict__ ef, const float* __restrict__ in0) {
    int i = blockIdx.x * 256 + threadIdx.x;
    if (i < BATCH * FEAT * NSQ) {
        int m = i & 127;
        int n = (i >> 7) & 127;
        int b = i >> 23;
        ef[i] *= in0[(long)b * CIN0 * NSQ + n * NN + m];
    }
}

__global__ void sums_kernel(const float* __restrict__ ef,
                            float* __restrict__ rowsum, float* __restrict__ colsum) {
    int bc = blockIdx.x;          // B*FEAT blocks
    int t  = threadIdx.x;         // 128
    const float* p = ef + (long)bc * NSQ;
    float rs = 0.0f, cs = 0.0f;
    for (int m = 0; m < NN; ++m) { rs += p[t * NN + m]; cs += p[m * NN + t]; }
    rowsum[bc * NN + t] = rs;
    colsum[bc * NN + t] = cs;
}

__global__ void combine_kernel(float* __restrict__ ef,
                               const float* __restrict__ rowsum,
                               const float* __restrict__ colsum,
                               const float* __restrict__ eps) {
    int i = blockIdx.x * 256 + threadIdx.x;
    if (i < BATCH * FEAT * NSQ) {
        int m  = i & 127;
        int n  = (i >> 7) & 127;
        int bc = i >> 14;
        ef[i] = (eps[1] - 1.0f) * ef[i] + rowsum[bc * NN + n] + colsum[bc * NN + m];
    }
}

__global__ void pool_kernel(const float* __restrict__ x, float* __restrict__ pooled) {
    int bc = blockIdx.x;              // B*CH
    int b = bc >> 8, c = bc & 255;
    int t = threadIdx.x;              // 256
    const float* p = x + (long)bc * NSQ;
    float om = -1e30f, dm = -1e30f;
    for (int i = t; i < NSQ; i += 256) {
        int r = i >> 7, cc = i & 127;
        float v = p[i];
        if (r == cc) dm = fmaxf(dm, v); else om = fmaxf(om, v);
    }
    __shared__ float so[256], sd[256];
    so[t] = om; sd[t] = dm; __syncthreads();
    for (int s = 128; s > 0; s >>= 1) {
        if (t < s) { so[t] = fmaxf(so[t], so[t + s]); sd[t] = fmaxf(sd[t], sd[t + s]); }
        __syncthreads();
    }
    if (t == 0) { pooled[b * FEAT + c] = sd[0]; pooled[b * FEAT + CH + c] = so[0]; }
}

// ---------------------------------------------------------------------------
// MST (Prim) deaths -> h0 diagram
// ---------------------------------------------------------------------------
__global__ void mst_kernel(const float* __restrict__ ef, float* __restrict__ h0) {
    int g = blockIdx.x, t = threadIdx.x;   // 128 threads
    const float* f = ef + (long)g * NSQ;
    __shared__ float mind[NN]; __shared__ int vis[NN];
    __shared__ float rv[NN];   __shared__ int ri[NN];
    auto D = [&](int i, int j) {
        return 1.0f - 0.5f * (f[i * NN + j] + f[j * NN + i]);
    };
    vis[t] = (t == 0) ? 1 : 0;
    mind[t] = D(0, t);
    __syncthreads();
    for (int it = 0; it < NN - 1; ++it) {
        rv[t] = vis[t] ? 1e30f : mind[t]; ri[t] = t;
        __syncthreads();
        for (int s = 64; s > 0; s >>= 1) {
            if (t < s) {
                if (rv[t + s] < rv[t]) { rv[t] = rv[t + s]; ri[t] = ri[t + s]; }
            }
            __syncthreads();
        }
        int j = ri[0]; float d = rv[0];
        if (t == 0) {
            h0[(g * NN + it) * 2 + 0] = 0.0f;
            h0[(g * NN + it) * 2 + 1] = d;
            vis[j] = 1;
        }
        __syncthreads();
        mind[t] = fminf(mind[t], D(j, t));
        __syncthreads();
    }
    if (t == 0) {
        h0[(g * NN + NN - 1) * 2 + 0] = 0.0f;
        h0[(g * NN + NN - 1) * 2 + 1] = 1.0f;
    }
}

// ---------------------------------------------------------------------------
// Bitonic sort of the 8128 upper-tri distances -> h1 diagram (36 points)
// ---------------------------------------------------------------------------
__global__ __launch_bounds__(1024)
void sort_kernel(const float* __restrict__ ef, float* __restrict__ h1) {
    int g = blockIdx.x, t = threadIdx.x;
    __shared__ float s[SORTN];
    const float* f = ef + (long)g * NSQ;
    for (int idx = t; idx < NSQ; idx += 1024) {
        int i = idx >> 7, j = idx & 127;
        if (i < j) {
            int off = i * 127 - (i * (i - 1)) / 2 + (j - i - 1);
            s[off] = 1.0f - 0.5f * (f[i * NN + j] + f[j * NN + i]);
        }
    }
    if (t < SORTN - NPAIR) s[NPAIR + t] = 3.0e38f;
    __syncthreads();
    for (int k = 2; k <= SORTN; k <<= 1) {
        for (int j = k >> 1; j > 0; j >>= 1) {
            for (int i = t; i < SORTN; i += 1024) {
                int ixj = i ^ j;
                if (ixj > i) {
                    bool up = (i & k) == 0;
                    float a = s[i], b = s[ixj];
                    if ((a > b) == up) { s[i] = b; s[ixj] = a; }
                }
            }
            __syncthreads();
        }
    }
    if (t < K1V) {
        int idx = NN - 1 + 2 * t;
        h1[(g * 36 + t) * 2 + 0] = s[idx];
        h1[(g * 36 + t) * 2 + 1] = s[idx + 1];
    } else if (t < K1V + KE1V) {
        int r = t - K1V;
        h1[(g * 36 + t) * 2 + 0] = s[NPAIR - KE1V + r];
        h1[(g * 36 + t) * 2 + 1] = 1.0f;
    }
}

// ---------------------------------------------------------------------------
// PH branch: init FCs, 3x ISAB, PMA, dec, layer FCs  (one block per graph)
// ---------------------------------------------------------------------------
struct MabP {
    const float *Wq, *bq, *Wk, *bk, *Wv, *bv, *Wo, *bo;
    const float *g0, *be0, *g1, *be1;
};
struct BranchP {
    const float *W0, *b0, *W1, *b1;         // init fcs (2->16, 16->16)
    const float *I0, *I1, *I2;              // inducing points (50x16)
    MabP m[6];                              // isab i: m[2i]=mab0, m[2i+1]=mab1
    const float *S;                         // PMA seed (1x16)
    MabP pm;                                // PMA mab
    const float *Wd, *bd;                   // dec (16->16, no act)
    const float *Wl0, *bl0, *Wl1, *bl1;     // layers (32->64 relu, 64->200 relu)
};

__device__ void mab_run(int nq, float (*Q)[PH], int nk, float (*Ks)[PH],
                        const MabP& mp, float (*Out)[PH],
                        float (*Kt)[PH], float (*Vt)[PH]) {
    int t = threadIdx.x;
    if (t < nk) {
#pragma unroll
        for (int o = 0; o < PH; ++o) {
            float sk = mp.bk[o], sv = mp.bv[o];
            for (int d = 0; d < PH; ++d) {
                float x = Ks[t][d];
                sk += x * mp.Wk[d * PH + o];
                sv += x * mp.Wv[d * PH + o];
            }
            Kt[t][o] = sk; Vt[t][o] = sv;
        }
    }
    __syncthreads();
    float o16[PH];
    if (t < nq) {
        float q[PH];
#pragma unroll
        for (int o = 0; o < PH; ++o) {
            float s = mp.bq[o];
            for (int d = 0; d < PH; ++d) s += Q[t][d] * mp.Wq[d * PH + o];
            q[o] = s;
        }
        for (int h = 0; h < 4; ++h) {
            float mx = -1e30f;
            for (int kk = 0; kk < nk; ++kk) {
                float s = 0.0f;
#pragma unroll
                for (int d = 0; d < 4; ++d) s += q[h * 4 + d] * Kt[kk][h * 4 + d];
                s *= 0.25f;                       // 1/sqrt(16)
                if (s > mx) mx = s;
            }
            float se = 0.0f, oa0 = 0, oa1 = 0, oa2 = 0, oa3 = 0;
            for (int kk = 0; kk < nk; ++kk) {
                float s = 0.0f;
#pragma unroll
                for (int d = 0; d < 4; ++d) s += q[h * 4 + d] * Kt[kk][h * 4 + d];
                float w = __expf(s * 0.25f - mx);
                se += w;
                oa0 += w * Vt[kk][h * 4 + 0];
                oa1 += w * Vt[kk][h * 4 + 1];
                oa2 += w * Vt[kk][h * 4 + 2];
                oa3 += w * Vt[kk][h * 4 + 3];
            }
            float inv = 1.0f / se;
            o16[h * 4 + 0] = q[h * 4 + 0] + oa0 * inv;
            o16[h * 4 + 1] = q[h * 4 + 1] + oa1 * inv;
            o16[h * 4 + 2] = q[h * 4 + 2] + oa2 * inv;
            o16[h * 4 + 3] = q[h * 4 + 3] + oa3 * inv;
        }
        // layernorm 0
        float mn = 0.0f;
#pragma unroll
        for (int o = 0; o < PH; ++o) mn += o16[o];
        mn *= (1.0f / PH);
        float var = 0.0f;
#pragma unroll
        for (int o = 0; o < PH; ++o) { float d = o16[o] - mn; var += d * d; }
        var *= (1.0f / PH);
        float rs = rsqrtf(var + 1e-5f);
#pragma unroll
        for (int o = 0; o < PH; ++o) o16[o] = (o16[o] - mn) * rs * mp.g0[o] + mp.be0[o];
        // residual relu fc
        float r[PH];
#pragma unroll
        for (int o = 0; o < PH; ++o) {
            float s = mp.bo[o];
            for (int d = 0; d < PH; ++d) s += o16[d] * mp.Wo[d * PH + o];
            r[o] = fmaxf(s, 0.0f);
        }
#pragma unroll
        for (int o = 0; o < PH; ++o) o16[o] += r[o];
        // layernorm 1
        mn = 0.0f;
#pragma unroll
        for (int o = 0; o < PH; ++o) mn += o16[o];
        mn *= (1.0f / PH);
        var = 0.0f;
#pragma unroll
        for (int o = 0; o < PH; ++o) { float d = o16[o] - mn; var += d * d; }
        var *= (1.0f / PH);
        rs = rsqrtf(var + 1e-5f);
#pragma unroll
        for (int o = 0; o < PH; ++o) o16[o] = (o16[o] - mn) * rs * mp.g1[o] + mp.be1[o];
    }
    __syncthreads();
    if (t < nq) {
#pragma unroll
        for (int o = 0; o < PH; ++o) Out[t][o] = o16[o];
    }
    __syncthreads();
}

__global__ __launch_bounds__(128)
void ph_branch_kernel(const float* __restrict__ hpts, int npts, int hstride,
                      BranchP bp, float* __restrict__ out, int outStride) {
    int g = blockIdx.x, t = threadIdx.x;
    __shared__ float Eb[NN][PH], Tb[NN][PH], Kt[NN][PH], Vt[NN][PH];
    __shared__ float Hb[56][PH];
    __shared__ float ssum[PH], fvec[32], f64[64];
    const float* h = hpts + (long)g * hstride;

    if (t < npts) {
        float x0 = h[t * 2 + 0], x1 = h[t * 2 + 1];
        float a[PH];
#pragma unroll
        for (int o = 0; o < PH; ++o)
            a[o] = fmaxf(bp.b0[o] + x0 * bp.W0[o] + x1 * bp.W0[PH + o], 0.0f);
#pragma unroll
        for (int o = 0; o < PH; ++o) {
            float s = bp.b1[o];
            for (int d = 0; d < PH; ++d) s += a[d] * bp.W1[d * PH + o];
            Eb[t][o] = fmaxf(s, 0.0f);
        }
    }
    __syncthreads();
    if (t < PH) {
        float s = 0.0f;
        for (int r = 0; r < npts; ++r) s += Eb[r][t];
        ssum[t] = s;
    }
    __syncthreads();

    for (int ib = 0; ib < 3; ++ib) {
        const float* I = (ib == 0) ? bp.I0 : (ib == 1) ? bp.I1 : bp.I2;
        if (t < PHI) {
#pragma unroll
            for (int o = 0; o < PH; ++o) Tb[t][o] = I[t * PH + o];
        }
        __syncthreads();
        mab_run(PHI, Tb, npts, Eb, bp.m[2 * ib + 0], Hb, Kt, Vt);  // H = MAB(I, E)
        mab_run(npts, Eb, PHI, Hb, bp.m[2 * ib + 1], Tb, Kt, Vt);  // E' = MAB(E, H)
        if (t < npts) {
#pragma unroll
            for (int o = 0; o < PH; ++o) Eb[t][o] = Tb[t][o];
        }
        __syncthreads();
    }
    // PMA
    if (t == 0) {
#pragma unroll
        for (int o = 0; o < PH; ++o) Tb[0][o] = bp.S[o];
    }
    __syncthreads();
    mab_run(1, Tb, npts, Eb, bp.pm, Hb, Kt, Vt);                    // st = Hb[0]
    // dec (no act) + concat
    if (t < PH) {
        float s = bp.bd[t];
        for (int d = 0; d < PH; ++d) s += Hb[0][d] * bp.Wd[d * PH + t];
        fvec[PH + t] = s;
        fvec[t] = ssum[t];
    }
    __syncthreads();
    if (t < 64) {
        float s = bp.bl0[t];
        for (int d = 0; d < 32; ++d) s += fvec[d] * bp.Wl0[d * 64 + t];
        f64[t] = fmaxf(s, 0.0f);
    }
    __syncthreads();
    for (int o = t; o < 200; o += 128) {
        float s = bp.bl1[o];
        for (int d = 0; d < 64; ++d) s += f64[d] * bp.Wl1[d * 200 + o];
        out[(long)g * outStride + o] = fmaxf(s, 0.0f);
    }
}

// ---------------------------------------------------------------------------
// Head: ph_layers (relu, none) -> combine with pooled -> fc chain -> 10 logits
// ---------------------------------------------------------------------------
__global__ __launch_bounds__(256)
void head_kernel(const float* __restrict__ branch, const float* __restrict__ pooled,
                 const float* __restrict__ eps,
                 const float* Wp0, const float* bp0, const float* Wp1, const float* bp1,
                 const float* Wf0, const float* bf0, const float* Wf1, const float* bf1,
                 const float* Wf2, const float* bf2,
                 float* __restrict__ out) {
    int g = blockIdx.x, t = threadIdx.x;
    __shared__ float tot[400], pa[512], pb[512], h256[256];
    for (int i = t; i < 400; i += 256) tot[i] = branch[g * 400 + i];
    __syncthreads();
    for (int o = t; o < 512; o += 256) {
        float s = bp0[o];
        for (int d = 0; d < 400; ++d) s += tot[d] * Wp0[d * 512 + o];
        pa[o] = fmaxf(s, 0.0f);
    }
    __syncthreads();
    for (int o = t; o < 512; o += 256) {
        float s = bp1[o];
        for (int d = 0; d < 512; ++d) s += pa[d] * Wp1[d * 512 + o];
        // ph_layers[1] has act=False
        pb[o] = pooled[g * 512 + o] + (1.0f + eps[2]) * s;
    }
    __syncthreads();
    for (int o = t; o < 512; o += 256) {
        float s = bf0[o];
        for (int d = 0; d < 512; ++d) s += pb[d] * Wf0[d * 512 + o];
        pa[o] = fmaxf(s, 0.0f);
    }
    __syncthreads();
    for (int o = t; o < 256; o += 256) {
        float s = bf1[o];
        for (int d = 0; d < 512; ++d) s += pa[d] * Wf1[d * 256 + o];
        h256[o] = fmaxf(s, 0.0f);
    }
    __syncthreads();
    if (t < 10) {
        float s = bf2[t];
        for (int d = 0; d < 256; ++d) s += h256[d] * Wf2[d * 10 + t];
        out[g * 10 + t] = s;
    }
}

// ---------------------------------------------------------------------------
// Host side
// ---------------------------------------------------------------------------
static inline void launch_gemm(hipStream_t st, bool arow,
                               const float* A, const float* B0, const float* B1,
                               int Ksplit, const float* bias, float* C,
                               int M, int K, int P,
                               long sA, long sB0, long sB1, long sC,
                               int batch, int act) {
    dim3 g((P + TP - 1) / TP, (M + TM - 1) / TM, batch), b(128);
    if (arow)
        wmma_gemm<true><<<g, b, 0, st>>>(A, B0, B1, Ksplit, bias, C, M, K, P, sA, sB0, sB1, sC, act);
    else
        wmma_gemm<false><<<g, b, 0, st>>>(A, B0, B1, Ksplit, bias, C, M, K, P, sA, sB0, sB1, sC, act);
}

static MabP mk_mab(void* const* d_in, int i) {
    MabP m;
    m.Wq = (const float*)d_in[i + 0];  m.bq = (const float*)d_in[i + 1];
    m.Wk = (const float*)d_in[i + 2];  m.bk = (const float*)d_in[i + 3];
    m.Wv = (const float*)d_in[i + 4];  m.bv = (const float*)d_in[i + 5];
    m.Wo = (const float*)d_in[i + 6];  m.bo = (const float*)d_in[i + 7];
    m.g0 = (const float*)d_in[i + 8];  m.be0 = (const float*)d_in[i + 9];
    m.g1 = (const float*)d_in[i + 10]; m.be1 = (const float*)d_in[i + 11];
    return m;
}

static BranchP mk_branch(void* const* d_in, int initB, int encB, int pmaB,
                         int decB, int layB) {
    BranchP bp;
    bp.W0 = (const float*)d_in[initB + 0]; bp.b0 = (const float*)d_in[initB + 1];
    bp.W1 = (const float*)d_in[initB + 2]; bp.b1 = (const float*)d_in[initB + 3];
    bp.I0 = (const float*)d_in[encB + 0];
    bp.m[0] = mk_mab(d_in, encB + 1);
    bp.m[1] = mk_mab(d_in, encB + 13);
    bp.I1 = (const float*)d_in[encB + 25];
    bp.m[2] = mk_mab(d_in, encB + 26);
    bp.m[3] = mk_mab(d_in, encB + 38);
    bp.I2 = (const float*)d_in[encB + 50];
    bp.m[4] = mk_mab(d_in, encB + 51);
    bp.m[5] = mk_mab(d_in, encB + 63);
    bp.S  = (const float*)d_in[pmaB];
    bp.pm = mk_mab(d_in, pmaB + 1);
    bp.Wd = (const float*)d_in[decB + 0]; bp.bd = (const float*)d_in[decB + 1];
    bp.Wl0 = (const float*)d_in[layB + 0]; bp.bl0 = (const float*)d_in[layB + 1];
    bp.Wl1 = (const float*)d_in[layB + 2]; bp.bl1 = (const float*)d_in[layB + 3];
    return bp;
}

extern "C" void kernel_launch(void* const* d_in, const int* in_sizes, int n_in,
                              void* d_out, int out_size, void* d_ws, size_t ws_size,
                              hipStream_t stream) {
    (void)in_sizes; (void)n_in; (void)out_size; (void)ws_size;
    auto F = [&](int i) { return (const float*)d_in[i]; };
    const float* input = F(0);
    const float* eps   = F(1);
    float* out = (float*)d_out;
    float* pred = out + BATCH * 10;                    // pred_adjs region

    // Workspace layout
    float* ws = (float*)d_ws;
    const size_t SZ = (size_t)BATCH * CH * NSQ;        // 33.5M floats = 134 MB
    float* R0 = ws;
    float* R1 = ws + SZ;
    float* R2 = ws + 2 * SZ;
    float* R3 = ws + 3 * SZ;
    float* sm = ws + 4 * SZ;
    float* vdiag  = sm;  sm += (size_t)BATCH * CH * NN;
    float* spbuf  = sm;  sm += (size_t)BATCH * NN;
    float* rowsum = sm;  sm += (size_t)BATCH * FEAT * NN;
    float* colsum = sm;  sm += (size_t)BATCH * FEAT * NN;
    float* pooled = sm;  sm += (size_t)BATCH * FEAT;
    float* h0w    = sm;  sm += (size_t)BATCH * NN * 2;
    float* h1w    = sm;  sm += (size_t)BATCH * 36 * 2;
    float* branch = sm;  sm += (size_t)BATCH * 400;

    // ---- PPGN blocks ----
    // buffer rotation: x -> {b0: input, b1: R0, b2: R2}; xnew -> {R0, R2, R0}
    const float* xcur = input;
    float* m1buf[3] = { R2, R2, R0 };
    float* xnew[3]  = { R0, R2, R0 };
    for (int bl = 0; bl < 3; ++bl) {
        int base = 2 + bl * 10;
        int cin  = (bl == 0) ? CIN0 : CH;
        long sBx = (long)cin * NSQ;
        // mlp1
        launch_gemm(stream, false, F(base + 0), xcur, nullptr, cin, F(base + 1), R1,
                    CH, cin, NSQ, 0, sBx, 0, (long)CH * NSQ, BATCH, 1);
        launch_gemm(stream, false, F(base + 2), R1, nullptr, CH, F(base + 3), m1buf[bl],
                    CH, CH, NSQ, 0, (long)CH * NSQ, 0, (long)CH * NSQ, BATCH, 1);
        // mlp2
        launch_gemm(stream, false, F(base + 4), xcur, nullptr, cin, F(base + 5), R1,
                    CH, cin, NSQ, 0, sBx, 0, (long)CH * NSQ, BATCH, 1);
        launch_gemm(stream, false, F(base + 6), R1, nullptr, CH, F(base + 7), R3,
                    CH, CH, NSQ, 0, (long)CH * NSQ, 0, (long)CH * NSQ, BATCH, 1);
        // mult = m1 @ m2  (batched 128x128x128 over B*C)
        launch_gemm(stream, true, m1buf[bl], R3, nullptr, NN, nullptr, R1,
                    NN, NN, NN, (long)NSQ, (long)NSQ, 0, (long)NSQ, BATCH * CH, 0);
        // skip conv on concat(x, mult)  (no activation)
        launch_gemm(stream, false, F(base + 8), xcur, R1, cin, F(base + 9), xnew[bl],
                    CH, cin + CH, NSQ, 0, sBx, (long)CH * NSQ, (long)CH * NSQ, BATCH, 0);
        xcur = xnew[bl];
    }
    // xcur == R0 (final x)

    // ---- pooling (frees R0 afterwards) ----
    pool_kernel<<<BATCH * CH, 256, 0, stream>>>(R0, pooled);

    // ---- diag + sp_edge ----
    {
        int n = BATCH * CH * NN;
        diag_kernel<<<(n + 255) / 256, 256, 0, stream>>>(R0, vdiag);
        launch_gemm(stream, false, F(32), vdiag, nullptr, CH, F(33), spbuf,
                    1, CH, NN, 0, (long)CH * NN, 0, (long)NN, BATCH, 0);
        vsp_kernel<<<(n + 255) / 256, 256, 0, stream>>>(vdiag, spbuf, eps);
    }

    // ---- edge features ----
    float* ef0 = R2;   // 512ch, spans R2|R3 (contiguous)
    float* ef1 = R0;   // 512ch, spans R0|R1
    {
        int n = BATCH * FEAT * NSQ;
        build_ef_kernel<<<(n + 255) / 256, 256, 0, stream>>>(vdiag, ef0);
        launch_gemm(stream, false, F(34), ef0, nullptr, FEAT, F(35), ef1,
                    FEAT, FEAT, NSQ, 0, (long)FEAT * NSQ, 0, (long)FEAT * NSQ, BATCH, 1);
        mask_adj_kernel<<<(n + 255) / 256, 256, 0, stream>>>(ef1, input);
        sums_kernel<<<BATCH * FEAT, 128, 0, stream>>>(ef1, rowsum, colsum);
        combine_kernel<<<(n + 255) / 256, 256, 0, stream>>>(ef1, rowsum, colsum, eps);
        launch_gemm(stream, false, F(36), ef1, nullptr, FEAT, F(37), R2,
                    CH, FEAT, NSQ, 0, (long)FEAT * NSQ, 0, (long)CH * NSQ, BATCH, 1);
        launch_gemm(stream, false, F(38), R2, nullptr, CH, F(39), pred,
                    1, CH, NSQ, 0, (long)CH * NSQ, 0, (long)NSQ, BATCH, 2);
    }

    // ---- persistent homology diagrams ----
    mst_kernel<<<BATCH, NN, 0, stream>>>(pred, h0w);
    sort_kernel<<<BATCH, 1024, 0, stream>>>(pred, h1w);

    // ---- PH branches ----
    BranchP bp0 = mk_branch(d_in, 40, 48, 198, 224, 228);
    BranchP bp1 = mk_branch(d_in, 44, 123, 211, 226, 232);
    ph_branch_kernel<<<BATCH, 128, 0, stream>>>(h0w, NN, NN * 2, bp0, branch + 0, 400);
    ph_branch_kernel<<<BATCH, 128, 0, stream>>>(h1w, 36, 36 * 2, bp1, branch + 200, 400);

    // ---- head ----
    head_kernel<<<BATCH, 256, 0, stream>>>(branch, pooled, eps,
        F(236), F(237), F(238), F(239),
        F(240), F(241), F(242), F(243), F(244), F(245),
        out);
}
